// VectorQuantizer_37666863186435
// MI455X (gfx1250) — matrix operand order repaired and checked by reading, not compiled, not run on hardware
//
#include <hip/hip_runtime.h>

// MI455X / gfx1250 VQ kernel, v4.
// - cross-term GEMM (32768x1024x256) via v_wmma_f32_16x16x32_bf16, f32 accum.
// - codebook pre-converted to bf16 once (prep kernel).
// - codebook chunks staged to double-buffered LDS with GLOBAL_LOAD_ASYNC_TO_LDS_B128
//   (ASYNCcnt-tracked), overlapped with WMMA compute.
// - each wave owns TWO 16-row M tiles (A register-resident): every B fragment
//   fetched from LDS feeds two WMMAs.
// - B fragments triple-buffered with lookahead-2 across a flattened (s,k) loop
//   so ds_load latency is hidden behind WMMA (partial s_wait_dscnt).
// - argmin state packed as one u32 key per slot: order-preserving float map,
//   top 22 bits distance | low 10 bits index -> v_min_u32 running argmin,
//   ties -> smallest index (matches jnp.argmin).

typedef unsigned int  uint32;
typedef __bf16  v16bf __attribute__((ext_vector_type(16)));
typedef float   v8f   __attribute__((ext_vector_type(8)));

#define D_DIM       256
#define CHUNK_ROWS  64          // codebook rows staged in LDS per iteration
#define ROW_BYTES   528         // 256*2 bf16 + 16B pad (conflict-free b128 reads)
#define ROW_U4      33          // 528 / 16
#define CB_ROW_B    512         // packed bf16 codebook row bytes in workspace

__device__ __forceinline__ uint32 f2bf1(float f) {
    uint32 u = __builtin_bit_cast(uint32, f);
    u += 0x7FFFu + ((u >> 16) & 1u);          // round-to-nearest-even
    return u >> 16;
}
__device__ __forceinline__ uint32 pack2(float a, float b) {
    return f2bf1(a) | (f2bf1(b) << 16);
}

// dist = cs - 2*acc, mapped to order-preserving u32, index in low 10 bits
__device__ __forceinline__ uint32 makekey(float acc, float cs, uint32 nidx) {
    float d = __builtin_fmaf(-2.0f, acc, cs);
    uint32 b = __builtin_bit_cast(uint32, d);
    uint32 m = b ^ (0x80000000u | (uint32)((int)b >> 31));
    return (m & 0xFFFFFC00u) | nidx;
}

// 16B global -> LDS async copy (per lane). VDST = LDS byte address VGPR.
__device__ __forceinline__ void async_copy16(unsigned lds_off, const void* gptr) {
    unsigned long long ga = (unsigned long long)(size_t)gptr;
    asm volatile("global_load_async_to_lds_b128 %0, %1, off"
                 :: "v"(lds_off), "v"(ga) : "memory");
}

template <int N>
__device__ __forceinline__ void wait_async() {
#if __has_builtin(__builtin_amdgcn_s_wait_asynccnt)
    __builtin_amdgcn_s_wait_asynccnt(N);
#else
    asm volatile("s_wait_asynccnt %0" :: "i"(N) : "memory");
#endif
}

// ---------------- prep kernel: bf16 codebook + c_sq ---------------------------
__global__ void __launch_bounds__(256)
vq_prep_kernel(const float* __restrict__ cb,
               uint32* __restrict__ cb16,
               float* __restrict__ csq) {
    const int lane = threadIdx.x & 31;
    const int row  = (blockIdx.x * blockDim.x + threadIdx.x) >> 5;
    const float* src = cb + (size_t)row * D_DIM + lane * 8;
    float4 a = *(const float4*)(src);
    float4 b = *(const float4*)(src + 4);
    uint32* dst = cb16 + (size_t)row * (CB_ROW_B / 4) + lane * 4;
    dst[0] = pack2(a.x, a.y);
    dst[1] = pack2(a.z, a.w);
    dst[2] = pack2(b.x, b.y);
    dst[3] = pack2(b.z, b.w);
    float s = a.x*a.x + a.y*a.y + a.z*a.z + a.w*a.w
            + b.x*b.x + b.y*b.y + b.z*b.z + b.w*b.w;
    #pragma unroll
    for (int m = 16; m >= 1; m >>= 1) s += __shfl_xor(s, m, 32);
    if (lane == 0) csq[row] = s;
}

// ---------------- main kernel ------------------------------------------------
// block = 256 threads = 8 waves; wave w handles M rows [blk*256 + 32w, +32)
__global__ void __launch_bounds__(256)
vq_main_kernel(const float* __restrict__ z,
               const float* __restrict__ cb,
               const uint32* __restrict__ cb16,
               const float* __restrict__ csq,
               float* __restrict__ zq,
               int*   __restrict__ idx_out,
               int K) {
    __shared__ uint4 lds[2][CHUNK_ROWS * ROW_U4];        // 2 x 33792 B double buffer

    const int tid    = threadIdx.x;
    const int lane   = tid & 31;
    const int half   = lane >> 4;                        // 0: lanes 0-15, 1: lanes 16-31
    const int nlane  = lane & 15;
    const int wave   = tid >> 5;
    const int m_base = blockIdx.x * 256 + wave * 32;

    // LDS byte offsets of the two staging buffers + this thread's slice
    const int srow = tid >> 2;                           // 0..63
    const int sq   = tid & 3;
    const unsigned buf_off[2] = {
        (unsigned)(size_t)(&lds[0][0]) + (unsigned)(srow * ROW_BYTES + sq * 128),
        (unsigned)(size_t)(&lds[1][0]) + (unsigned)(srow * ROW_BYTES + sq * 128)
    };
    const char* cb16b = (const char*)cb16;

    // ---- load the wave's two 16x256 z tiles as bf16 A-fragments (in regs) ----
    union AF { v16bf bf; uint32 u[8]; };
    AF afrag[2][8];
    #pragma unroll
    for (int t = 0; t < 2; ++t) {
        const float* zrow = z + (size_t)(m_base + t * 16 + nlane) * D_DIM;
        #pragma unroll
        for (int k = 0; k < 8; ++k) {
            int d0 = k * 32 + half * 16;
            float4 a0 = *(const float4*)(zrow + d0);
            float4 a1 = *(const float4*)(zrow + d0 + 4);
            float4 a2 = *(const float4*)(zrow + d0 + 8);
            float4 a3 = *(const float4*)(zrow + d0 + 12);
            afrag[t][k].u[0] = pack2(a0.x, a0.y);
            afrag[t][k].u[1] = pack2(a0.z, a0.w);
            afrag[t][k].u[2] = pack2(a1.x, a1.y);
            afrag[t][k].u[3] = pack2(a1.z, a1.w);
            afrag[t][k].u[4] = pack2(a2.x, a2.y);
            afrag[t][k].u[5] = pack2(a2.z, a2.w);
            afrag[t][k].u[6] = pack2(a3.x, a3.y);
            afrag[t][k].u[7] = pack2(a3.z, a3.w);
        }
    }

    uint32 key[2][8];
    #pragma unroll
    for (int t = 0; t < 2; ++t)
        #pragma unroll
        for (int r = 0; r < 8; ++r) key[t][r] = 0xFFFFFFFFu;

    const int NCH = K / CHUNK_ROWS;                      // 16
    const v8f vzero = {};

    // ---- prologue: async-stage chunk 0 into buffer 0 ----
    {
        const char* src = cb16b + (size_t)srow * CB_ROW_B + sq * 128;
        #pragma unroll
        for (int j = 0; j < 8; ++j)
            async_copy16(buf_off[0] + j * 16, src + j * 16);
    }

    // ---- double-buffered pipeline over 64-row codebook chunks ----
    for (int i = 0; i < NCH; ++i) {
        __syncthreads();                                 // buf[(i+1)&1] readers (iter i-1) done
        if (i + 1 < NCH) {
            const char* src = cb16b + (size_t)(((i + 1) * CHUNK_ROWS) + srow) * CB_ROW_B
                            + sq * 128;
            const unsigned dst = buf_off[(i + 1) & 1];
            #pragma unroll
            for (int j = 0; j < 8; ++j)
                async_copy16(dst + j * 16, src + j * 16);
            wait_async<8>();                             // chunk i landed (in-order retire)
        } else {
            wait_async<0>();
        }
        __syncthreads();                                 // chunk i visible to all waves

        const int c = i * CHUNK_ROWS;
        // B 32x16 bf16 layout: lane holds col N=lane%16, K = 16*half + 0..15
        const uint4* base = &lds[i & 1][0] + nlane * ROW_U4 + half * 2;

        float cs[4];
        #pragma unroll
        for (int s = 0; s < 4; ++s) cs[s] = csq[c + s * 16 + nlane];

        // flattened (s,k) loop, triple-buffered B with lookahead-2
        union BF { v16bf bf; uint4 q[2]; };
        BF bfb[3];
        #pragma unroll
        for (int p = 0; p < 2; ++p) {                    // prefetch idx 0,1 (s=0)
            const uint4* q = base + (p & 7) * 4;
            bfb[p].q[0] = q[0];
            bfb[p].q[1] = q[1];
        }
        v8f acc0 = vzero;
        v8f acc1 = vzero;
        #pragma unroll
        for (int idx = 0; idx < 32; ++idx) {
            const int k = idx & 7;
            const int s = idx >> 3;
            if (idx + 2 < 32) {
                const int nn = idx + 2;
                const uint4* q = base + (nn >> 3) * (16 * ROW_U4) + (nn & 7) * 4;
                bfb[nn % 3].q[0] = q[0];
                bfb[nn % 3].q[1] = q[1];
            }
            acc0 = __builtin_amdgcn_wmma_f32_16x16x32_bf16(
                    false, afrag[0][k].bf, false, bfb[idx % 3].bf,
                    (short)0, acc0, false, false);
            acc1 = __builtin_amdgcn_wmma_f32_16x16x32_bf16(
                    false, afrag[1][k].bf, false, bfb[idx % 3].bf,
                    (short)0, acc1, false, false);
            if (k == 7) {
                const uint32 nidx = (uint32)(c + s * 16 + nlane);
                #pragma unroll
                for (int r = 0; r < 8; ++r) {
                    uint32 c0 = makekey(acc0[r], cs[s], nidx);
                    if (c0 < key[0][r]) key[0][r] = c0;
                    uint32 c1 = makekey(acc1[r], cs[s], nidx);
                    if (c1 < key[1][r]) key[1][r] = c1;
                }
                acc0 = vzero;
                acc1 = vzero;
            }
        }
    }

    // ---- argmin reduction across the 16 lanes of each half ----
    // C layout: VGPR r holds (M = r + 8*half, N = lane%16)
    #pragma unroll
    for (int t = 0; t < 2; ++t) {
        #pragma unroll
        for (int r = 0; r < 8; ++r) {
            uint32 kk = key[t][r];
            #pragma unroll
            for (int m = 8; m >= 1; m >>= 1) {
                uint32 ok = (uint32)__shfl_xor((int)kk, m, 16);
                if (ok < kk) kk = ok;
            }
            key[t][r] = kk;
        }
    }

    // ---- write indices + gather winning codebook rows into z_q ----
    #pragma unroll
    for (int i = 0; i < 32; ++i) {
        const int t    = i >> 4;
        const int slot = i & 7;
        const int src  = (i & 8) ? 16 : 0;
        int idx_i = (int)((uint32)__shfl((int)key[t][slot], src, 32) & 1023u);
        int m     = m_base + i;
        if (lane == (i & 31)) idx_out[m] = idx_i;
        const float4* sp = (const float4*)(cb + (size_t)idx_i * D_DIM);
        float4*       dp = (float4*)(zq + (size_t)m * D_DIM);
        dp[lane]      = sp[lane];                        // 64 float4 per row,
        dp[lane + 32] = sp[lane + 32];                   // 2 per lane
    }
}

extern "C" void kernel_launch(void* const* d_in, const int* in_sizes, int n_in,
                              void* d_out, int out_size, void* d_ws, size_t ws_size,
                              hipStream_t stream) {
    const float* z  = (const float*)d_in[0];
    const float* cb = (const float*)d_in[1];

    const int M = in_sizes[0] / D_DIM;                   // 8*4096 = 32768
    const int K = in_sizes[1] / D_DIM;                   // 1024

    float* zq      = (float*)d_out;
    int*   idx_out = (int*)((float*)d_out + (size_t)M * D_DIM);

    float*  csq  = (float*)d_ws;                         // K floats
    uint32* cb16 = (uint32*)((char*)d_ws + 4096);        // K * 512 B packed bf16

    vq_prep_kernel<<<K * 32 / 256, 256, 0, stream>>>(cb, cb16, csq);
    vq_main_kernel<<<M / 256, 256, 0, stream>>>(z, cb, cb16, csq, zq, idx_out, K);
}
